// SSMTranslator_24060406792345
// MI455X (gfx1250) — compile-verified, matrix operand order checked by reading, not run on hardware
//
#include <hip/hip_runtime.h>

#define USE_ASYNC 1   // CDNA5 global_load_async_to_lds_b128 path (fallback: uint4 copy)

// ---------------- model constants ----------------
#define BSZ     2
#define TENC    256
#define TDEC    128
#define DMODEL  1024
#define NH      16
#define CHD     64
#define NENC    64
#define NDEC    8
#define LL      8
#define KK      4
#define VOCABN  20000
#define CH_ENC  192   // CHD + 2*NENC
#define CH_ENCL 128   // CHD + NENC
#define CH_DEC  80    // CHD + 2*NDEC

typedef __attribute__((ext_vector_type(16))) __bf16 v16bf;
typedef __attribute__((ext_vector_type(8)))  float  v8f;

// ---------------- WMMA helpers (CDNA5 bf16, f32 accum) ----------------
__device__ inline v8f wmma_bf(v16bf a, v16bf b, v8f c) {
  // D = A(16x32) * B(32x16) + C  -> v_wmma_f32_16x16x32_bf16
  return __builtin_amdgcn_wmma_f32_16x16x32_bf16(false, a, false, b, (short)0, c,
                                                 false, false);
}

// 16-byte global -> LDS copy. Async path: per-lane DMA to LDS, ASYNCcnt-tracked.
__device__ inline void lds_cp16(__bf16* l, const __bf16* g) {
#if USE_ASYNC
  unsigned loff = (unsigned)(size_t)l;                 // LDS byte offset = addr[31:0]
  unsigned long long ga = (unsigned long long)(size_t)g;
  asm volatile("global_load_async_to_lds_b128 %0, %1, off"
               :: "v"(loff), "v"(ga) : "memory");
#else
  *(uint4*)l = *(const uint4*)g;
#endif
}
__device__ inline void async_wait0() {
#if USE_ASYNC
  asm volatile("s_wait_asynccnt 0" ::: "memory");
#endif
}

// Matrix fragment, 16 "major" rows x 32 K, from LDS (row-major, stride ld).
// Used for WMMA A operands (major = M row) AND B operands staged N-major
// (major = N column): lane l holds major (l&15); K-group (l>>4)*8;
// vgpr j -> K = (j>>2)*16 + (j&3)*2  (pairs loaded as one 32-bit word).
__device__ inline v16bf ld_frag(const __bf16* base, int ld, int row0, int k0) {
  int lane = threadIdx.x & 31;
  const __bf16* p = base + (row0 + (lane & 15)) * ld + k0 + ((lane >> 4) << 3);
  union { v16bf v; unsigned u[8]; } r;
#pragma unroll
  for (int j = 0; j < 8; ++j) {
    int k = ((j >> 2) << 4) + ((j & 3) << 1);
    r.u[j] = *(const unsigned*)(p + k);
  }
  return r.v;
}

__device__ inline float silu_f(float x) { return x / (1.f + __expf(-x)); }

// ---------------- generic batched WMMA GEMM ----------------
// C[z] = A[z](MxK,row-major) * B[z](NxK,row-major i.e. N-major) + bias + res.
// Block tile 64x128, 8 waves.  REQUIRES: M % 64 == 0, K % 32 == 0.
// LDS row strides are multiples of 8 elems so all 16B stores are aligned.
__global__ __launch_bounds__(256) void k_gemm(
    const __bf16* __restrict__ A, int lda, long long sA1, long long sA2,
    const __bf16* __restrict__ B, int ldb, long long sB1, long long sB2,
    float* C, __bf16* Cbf, int ldc, long long sC1, long long sC2,
    const float* bias, long long sBias2, const float* res,
    int M, int N, int K, int ntN, int zdiv) {
  __shared__ __bf16 As[64][40];
  __shared__ __bf16 BsT[128][40];
  int z = blockIdx.y;
  int z1 = z / zdiv, z2 = z % zdiv;
  const __bf16* Ap = A + z1 * sA1 + z2 * sA2;
  const __bf16* Bp = B + z1 * sB1 + z2 * sB2;
  long long coff = z1 * sC1 + z2 * sC2;
  int tM = (blockIdx.x / ntN) * 64;
  int tN = (blockIdx.x % ntN) * 128;
  int tid = threadIdx.x, lane = tid & 31, wave = tid >> 5;
  int wm = (wave >> 2) * 32, wn = (wave & 3) * 32;
  v8f z8 = {0.f, 0.f, 0.f, 0.f, 0.f, 0.f, 0.f, 0.f};
  v8f acc00 = z8, acc01 = z8, acc10 = z8, acc11 = z8;
  int arow = tid >> 2, acol = (tid & 3) * 8;
  for (int k0 = 0; k0 < K; k0 += 32) {
    // A tile 64(M)x32(K): one 16B chunk per thread
    lds_cp16(&As[arow][acol], Ap + (long long)(tM + arow) * lda + k0 + acol);
    // B tile 128(N)x32(K), N-major: two 16B chunks per thread
#pragma unroll
    for (int q = 0; q < 2; ++q) {
      int cid = tid + q * 256;
      int rn = cid >> 2, ck = (cid & 3) * 8;
      int gn = tN + rn;
      if (gn < N)
        lds_cp16(&BsT[rn][ck], Bp + (long long)gn * ldb + k0 + ck);
      else
        *(uint4*)&BsT[rn][ck] = make_uint4(0u, 0u, 0u, 0u);
    }
    async_wait0();
    __syncthreads();
    v16bf af0 = ld_frag(&As[0][0], 40, wm, 0);
    v16bf af1 = ld_frag(&As[0][0], 40, wm + 16, 0);
    v16bf bf0 = ld_frag(&BsT[0][0], 40, wn, 0);
    v16bf bf1 = ld_frag(&BsT[0][0], 40, wn + 16, 0);
    acc00 = wmma_bf(af0, bf0, acc00);
    acc01 = wmma_bf(af0, bf1, acc01);
    acc10 = wmma_bf(af1, bf0, acc10);
    acc11 = wmma_bf(af1, bf1, acc11);
    __syncthreads();
  }
  int rh = (lane >> 4) << 3, rl = lane & 15;
#pragma unroll
  for (int i = 0; i < 2; ++i)
#pragma unroll
    for (int j = 0; j < 2; ++j) {
      v8f a = (i == 0) ? (j == 0 ? acc00 : acc01) : (j == 0 ? acc10 : acc11);
#pragma unroll
      for (int v = 0; v < 8; ++v) {
        int r = tM + wm + i * 16 + v + rh;
        int c = tN + wn + j * 16 + rl;
        if (c < N) {
          float val = a[v];
          if (bias) val += bias[z2 * sBias2 + c];
          long long o = coff + (long long)r * ldc + c;
          if (res) val += res[o];
          if (C) C[o] = val;
          if (Cbf) Cbf[o] = (__bf16)val;
        }
      }
    }
}

// ---------------- fused masked-linear-attention SSM (encoder) ----------------
// per (b,h): S = C*Bm^T (decay-masked), Y = S*X ; flash-style over 64-row tiles.
// Cbf  : (z, T, 64)  A-operand (row-major M x K)
// Bmbf : (z, T, 64)  first-GEMM B operand, already N(j)-major
// XbfT : (z, 64, T)  second-GEMM B operand, N(c)-major
__global__ __launch_bounds__(256) void k_attn(const __bf16* __restrict__ Cbf,
                                              const __bf16* __restrict__ Bmbf,
                                              const __bf16* __restrict__ XbfT,
                                              const float* __restrict__ s,
                                              float* __restrict__ Y) {
  int z = blockIdx.y, b = z / NH, h = z % NH;
  int t0 = blockIdx.x * 64;
  const __bf16* Cp = Cbf + (long long)z * TENC * CHD;
  const __bf16* Bmp = Bmbf + (long long)z * TENC * CHD;
  const __bf16* XTp = XbfT + (long long)z * CHD * TENC;
  const float* sp = s + z * TENC;
  __shared__ __bf16 Cs[64][72];
  __shared__ __bf16 Ss[64][72];
  __shared__ __bf16 Ts[64][72];
  __shared__ float st[64], sj[64];
  int tid = threadIdx.x, lane = tid & 31, wave = tid >> 5;
  int wm = (wave >> 1) * 16, wn = (wave & 1) * 32;
  // 64x64 tile copy: 512 16B-chunks, 2 per thread
#pragma unroll
  for (int q = 0; q < 2; ++q) {
    int cid = tid + q * 256;
    int r = cid >> 3, c = (cid & 7) * 8;
    lds_cp16(&Cs[r][c], Cp + (long long)(t0 + r) * CHD + c);
  }
  if (tid < 64) st[tid] = sp[t0 + tid];
  async_wait0();
  __syncthreads();
  v16bf aC0 = ld_frag(&Cs[0][0], 72, wm, 0);
  v16bf aC1 = ld_frag(&Cs[0][0], 72, wm, 32);
  v8f z8 = {0.f, 0.f, 0.f, 0.f, 0.f, 0.f, 0.f, 0.f};
  v8f Y0 = z8, Y1 = z8;
  int rh = (lane >> 4) << 3, rl = lane & 15;
  for (int jt = 0; jt <= (int)blockIdx.x; ++jt) {
    int j0 = jt * 64;
    __syncthreads();  // protect Ts (read as X last iter) before overwrite
    // Bm tile: rows = j (N index), cols = n (K index)
#pragma unroll
    for (int q = 0; q < 2; ++q) {
      int cid = tid + q * 256;
      int r = cid >> 3, c = (cid & 7) * 8;
      lds_cp16(&Ts[r][c], Bmp + (long long)(j0 + r) * CHD + c);
    }
    if (tid < 64) sj[tid] = sp[j0 + tid];
    async_wait0();
    __syncthreads();
    v8f S0 = z8, S1 = z8;
    S0 = wmma_bf(aC0, ld_frag(&Ts[0][0], 72, wn, 0), S0);
    S0 = wmma_bf(aC1, ld_frag(&Ts[0][0], 72, wn, 32), S0);
    S1 = wmma_bf(aC0, ld_frag(&Ts[0][0], 72, wn + 16, 0), S1);
    S1 = wmma_bf(aC1, ld_frag(&Ts[0][0], 72, wn + 16, 32), S1);
#pragma unroll
    for (int v = 0; v < 8; ++v) {
      int tr = wm + v + rh, tg = t0 + tr;
      int jc0 = wn + rl, jc1 = wn + 16 + rl;
      float v0 = (j0 + jc0 <= tg) ? S0[v] * __expf(st[tr] - sj[jc0]) : 0.f;
      float v1 = (j0 + jc1 <= tg) ? S1[v] * __expf(st[tr] - sj[jc1]) : 0.f;
      Ss[tr][jc0] = (__bf16)v0;
      Ss[tr][jc1] = (__bf16)v1;
    }
    __syncthreads();
    // X tile: rows = c (N index), cols = j (K index)  [from transposed X]
#pragma unroll
    for (int q = 0; q < 2; ++q) {
      int cid = tid + q * 256;
      int r = cid >> 3, c = (cid & 7) * 8;
      lds_cp16(&Ts[r][c], XTp + (long long)r * TENC + j0 + c);
    }
    async_wait0();
    __syncthreads();
    v16bf aS0 = ld_frag(&Ss[0][0], 72, wm, 0);
    v16bf aS1 = ld_frag(&Ss[0][0], 72, wm, 32);
    Y0 = wmma_bf(aS0, ld_frag(&Ts[0][0], 72, wn, 0), Y0);
    Y0 = wmma_bf(aS1, ld_frag(&Ts[0][0], 72, wn, 32), Y0);
    Y1 = wmma_bf(aS0, ld_frag(&Ts[0][0], 72, wn + 16, 0), Y1);
    Y1 = wmma_bf(aS1, ld_frag(&Ts[0][0], 72, wn + 16, 32), Y1);
  }
#pragma unroll
  for (int v = 0; v < 8; ++v) {
    int t = t0 + wm + v + rh;
    long long base = (((long long)(b * TENC + t)) * NH + h) * CHD;
    Y[base + wn + rl]      = Y0[v];
    Y[base + wn + 16 + rl] = Y1[v];
  }
}

// ---------------- small kernels ----------------
__global__ void k_cvt(const float* __restrict__ in, __bf16* __restrict__ out,
                      long long total) {
  long long i = (long long)blockIdx.x * blockDim.x + threadIdx.x;
  if (i < total) out[i] = (__bf16)in[i];
}

__global__ void k_embed(const int* __restrict__ ids, const float* __restrict__ E,
                        float* __restrict__ y, __bf16* ybf) {
  int tok = blockIdx.x;
  long long row = ids[tok];
  const float* src = E + row * DMODEL;
  for (int d = threadIdx.x; d < DMODEL; d += blockDim.x) {
    float v = src[d];
    y[(long long)tok * DMODEL + d] = v;
    if (ybf) ybf[(long long)tok * DMODEL + d] = (__bf16)v;
  }
}

__global__ void k_zero(float* p, int n) {
  int i = blockIdx.x * blockDim.x + threadIdx.x;
  if (i < n) p[i] = 0.f;
}

__global__ void k_logA(const float* __restrict__ y, const float* __restrict__ w,
                       const float* __restrict__ b, float* __restrict__ logA,
                       int total) {
  int i = blockIdx.x * blockDim.x + threadIdx.x;
  if (i >= total) return;
  int h = i % NH;
  long long bt = i / NH;
  const float* x = y + bt * DMODEL + h * CHD;
  const float* wp = w + h * CHD;
  float acc = b[h];
  for (int k = 0; k < CHD; ++k) acc += wp[k] * x[k];
  logA[i] = acc;
}

__global__ void k_cumsum(const float* __restrict__ logA, float* __restrict__ s) {
  int z = threadIdx.x;
  if (z >= BSZ * NH) return;
  int b = z / NH, h = z % NH;
  float run = 0.f;
  for (int t = 0; t < TENC; ++t) {
    run += logA[((long long)(b * TENC + t)) * NH + h];
    s[z * TENC + t] = run;
  }
}

__global__ void k_conv_silu(const float* __restrict__ xbc,
                            const float* __restrict__ cw,
                            const float* __restrict__ cb, float* __restrict__ XBC,
                            __bf16* XbfT, __bf16* Bmbf, __bf16* Cbf, int CH,
                            int hasC, long long total) {
  long long i = (long long)blockIdx.x * 256 + threadIdx.x;
  if (i >= total) return;
  int ch = (int)(i % CH);
  long long r = i / CH;
  int h = (int)(r % NH);
  long long bt = r / NH;
  int t = (int)(bt % TENC), b = (int)(bt / TENC);
  const float* wp = cw + ((long long)(h * CH + ch)) * KK;
  float acc = cb[h * CH + ch];
#pragma unroll
  for (int k = 0; k < KK; ++k) {
    int tt = t - (KK - 1) + k;
    if (tt >= 0)
      acc += xbc[(((long long)(b * TENC + tt)) * NH + h) * CH + ch] * wp[k];
  }
  float sv = silu_f(acc);
  XBC[i] = sv;
  int z = b * NH + h;
  if (ch < CHD)
    XbfT[((long long)z * CHD + ch) * TENC + t] = (__bf16)sv;       // X transposed
  else if (ch < CHD + NENC)
    Bmbf[((long long)z * TENC + t) * CHD + (ch - CHD)] = (__bf16)sv;  // Bm natural
  else if (hasC)
    Cbf[((long long)z * TENC + t) * CHD + (ch - CHD - NENC)] = (__bf16)sv;
}

__global__ void k_hT(const float* __restrict__ XBC, const float* __restrict__ s,
                     float* __restrict__ hs, int CH) {
  int z = blockIdx.y;
  int idx = blockIdx.x * 256 + threadIdx.x;  // 0..4095
  int c = idx >> 6, n = idx & 63;
  int b = z / NH, h = z % NH;
  float slast = s[z * TENC + TENC - 1];
  float acc = 0.f;
  for (int j = 0; j < TENC; ++j) {
    long long base = (((long long)(b * TENC + j)) * NH + h) * CH;
    float w = __expf(slast - s[z * TENC + j]);
    acc += w * XBC[base + c] * XBC[base + CHD + n];
  }
  hs[((long long)z * CHD + c) * NENC + n] = acc;
}

__global__ __launch_bounds__(256) void k_gate_ln(const float* __restrict__ g,
                                                 const float* __restrict__ Y,
                                                 const float* __restrict__ ng,
                                                 const float* __restrict__ nb,
                                                 __bf16* __restrict__ Ynbf) {
  long long bt = blockIdx.x;
  __shared__ float red[256];
  float vals[4];
  float sum = 0.f;
#pragma unroll
  for (int e = 0; e < 4; ++e) {
    int d = threadIdx.x + e * 256;
    float v = silu_f(g[bt * DMODEL + d]) * Y[bt * DMODEL + d];
    vals[e] = v;
    sum += v;
  }
  red[threadIdx.x] = sum;
  __syncthreads();
  for (int sft = 128; sft > 0; sft >>= 1) {
    if (threadIdx.x < sft) red[threadIdx.x] += red[threadIdx.x + sft];
    __syncthreads();
  }
  float mean = red[0] / DMODEL;
  __syncthreads();
  float sq = 0.f;
#pragma unroll
  for (int e = 0; e < 4; ++e) { float d = vals[e] - mean; sq += d * d; }
  red[threadIdx.x] = sq;
  __syncthreads();
  for (int sft = 128; sft > 0; sft >>= 1) {
    if (threadIdx.x < sft) red[threadIdx.x] += red[threadIdx.x + sft];
    __syncthreads();
  }
  float rstd = rsqrtf(red[0] / DMODEL + 1e-5f);
#pragma unroll
  for (int e = 0; e < 4; ++e) {
    int d = threadIdx.x + e * 256;
    Ynbf[bt * DMODEL + d] = (__bf16)((vals[e] - mean) * rstd * ng[d] + nb[d]);
  }
}

__global__ void k_ed(const float* __restrict__ hs, const float* __restrict__ ew,
                     const float* __restrict__ eb, float* __restrict__ Hs) {
  int idx = blockIdx.x * 256 + threadIdx.x;  // L*32*64*8
  int o = idx & 7, c = (idx >> 3) & 63, z = (idx >> 9) & 31, l = idx >> 14;
  const float* hp = hs + (((long long)l * 32 + z) * CHD + c) * NENC;
  const float* wp = ew + (l * NDEC + o) * NENC;
  float acc = eb[l * NDEC + o];
  for (int n = 0; n < NENC; ++n) acc += wp[n] * hp[n];
  Hs[(((long long)l * 32 + z) * CHD + c) * NDEC + o] = acc;
}

__global__ void k_copy_yt(const float* __restrict__ emb, float* __restrict__ yt,
                          int t) {
  int i = blockIdx.x * 256 + threadIdx.x;
  int b = i >> 10, d = i & 1023;
  yt[i] = emb[((long long)(b * TDEC + t)) * DMODEL + d];
}

__global__ void k_dec_proj(const float* __restrict__ yt,
                           const float* __restrict__ xw,
                           const float* __restrict__ xb,
                           const float* __restrict__ gw,
                           const float* __restrict__ gb,
                           const float* __restrict__ law,
                           const float* __restrict__ lab, float* xbc_dec,
                           float* gdec, float* adec, int l) {
  int z = blockIdx.x, b = z / NH, h = z % NH;
  __shared__ float xs[CHD];
  if (threadIdx.x < CHD) xs[threadIdx.x] = yt[b * DMODEL + h * CHD + threadIdx.x];
  __syncthreads();
  int j = threadIdx.x;
  if (j < CH_DEC) {
    const float* wp = xw + (((long long)l * NH + h) * CH_DEC + j) * CHD;
    float acc = xb[((long long)l * NH + h) * CH_DEC + j];
    for (int k = 0; k < CHD; ++k) acc += wp[k] * xs[k];
    xbc_dec[z * CH_DEC + j] = acc;
  } else if (j < CH_DEC + CHD) {
    int jj = j - CH_DEC;
    const float* wp = gw + (((long long)l * NH + h) * CHD + jj) * CHD;
    float acc = gb[((long long)l * NH + h) * CHD + jj];
    for (int k = 0; k < CHD; ++k) acc += wp[k] * xs[k];
    gdec[z * CHD + jj] = acc;
  } else if (j == CH_DEC + CHD) {
    const float* wp = law + ((long long)l * NH + h) * CHD;
    float acc = lab[l * NH + h];
    for (int k = 0; k < CHD; ++k) acc += wp[k] * xs[k];
    adec[z] = __expf(acc);
  }
}

__global__ __launch_bounds__(128) void k_dec_update(
    const float* __restrict__ xbc_dec, const float* __restrict__ cw,
    const float* __restrict__ cb, float* cache, float* Hs,
    const float* __restrict__ gdec, const float* __restrict__ adec, float* gY,
    int l) {
  int z = blockIdx.x, b = z / NH, h = z % NH;
  __shared__ float sx[CH_DEC];
  __shared__ float sA;
  int tid = threadIdx.x;
  if (tid < CH_DEC) {
    long long cbase =
        (((long long)l * BSZ + b) * (NH * CH_DEC) + h * CH_DEC + tid) * (KK - 1);
    float w0 = cache[cbase], w1 = cache[cbase + 1], w2 = cache[cbase + 2];
    float w3 = xbc_dec[z * CH_DEC + tid];
    long long widx = (long long)l * (NH * CH_DEC) + h * CH_DEC + tid;
    const float* wp = cw + widx * KK;
    float conv = w0 * wp[0] + w1 * wp[1] + w2 * wp[2] + w3 * wp[3] + cb[widx];
    sx[tid] = silu_f(conv);
    cache[cbase] = w1; cache[cbase + 1] = w2; cache[cbase + 2] = w3;
  }
  if (tid == 0) sA = adec[z];
  __syncthreads();
  if (tid < CHD) {
    float X = sx[tid];
    float* Hp = Hs + (((long long)l * 32 + z) * CHD + tid) * NDEC;
    float y = 0.f;
#pragma unroll
    for (int n = 0; n < NDEC; ++n) {
      float Hn = Hp[n] * sA + X * sx[CHD + n];
      Hp[n] = Hn;
      y += Hn * sx[CHD + NDEC + n];
    }
    gY[b * DMODEL + h * CHD + tid] = silu_f(gdec[z * CHD + tid]) * y;
  }
}

__global__ __launch_bounds__(256) void k_dec_ln(const float* __restrict__ gY,
                                                 const float* __restrict__ ng,
                                                 const float* __restrict__ nb,
                                                 float* __restrict__ Yn) {
  int b = blockIdx.x;
  __shared__ float red[256];
  float vals[4];
  float sum = 0.f;
#pragma unroll
  for (int e = 0; e < 4; ++e) {
    int d = threadIdx.x + e * 256;
    float v = gY[b * DMODEL + d];
    vals[e] = v;
    sum += v;
  }
  red[threadIdx.x] = sum;
  __syncthreads();
  for (int sft = 128; sft > 0; sft >>= 1) {
    if (threadIdx.x < sft) red[threadIdx.x] += red[threadIdx.x + sft];
    __syncthreads();
  }
  float mean = red[0] / DMODEL;
  __syncthreads();
  float sq = 0.f;
#pragma unroll
  for (int e = 0; e < 4; ++e) { float d = vals[e] - mean; sq += d * d; }
  red[threadIdx.x] = sq;
  __syncthreads();
  for (int sft = 128; sft > 0; sft >>= 1) {
    if (threadIdx.x < sft) red[threadIdx.x] += red[threadIdx.x + sft];
    __syncthreads();
  }
  float rstd = rsqrtf(red[0] / DMODEL + 1e-5f);
#pragma unroll
  for (int e = 0; e < 4; ++e) {
    int d = threadIdx.x + e * 256;
    Yn[b * DMODEL + d] = (vals[e] - mean) * rstd * ng[d] + nb[d];
  }
}

__global__ void k_dec_out(const float* __restrict__ Yn,
                          const float* __restrict__ ow,
                          const float* __restrict__ ob, float* yt,
                          __bf16* ytall, int l, int t, int last) {
  int i = blockIdx.x * 256 + threadIdx.x;  // B*1024
  int b = i >> 10, d = i & 1023;
  const float* wp = ow + ((long long)l * DMODEL + d) * DMODEL;
  const float* xp = Yn + b * DMODEL;
  float acc = ob[l * DMODEL + d];
  for (int k = 0; k < DMODEL; ++k) acc += wp[k] * xp[k];
  float nv = acc + yt[i];
  yt[i] = nv;
  if (last) ytall[((long long)(b * TDEC + t)) * DMODEL + d] = (__bf16)nv;
}

// ---------------- host orchestration ----------------
extern "C" void kernel_launch(void* const* d_in, const int* in_sizes, int n_in,
                              void* d_out, int out_size, void* d_ws,
                              size_t ws_size, hipStream_t stream) {
  (void)in_sizes; (void)n_in; (void)out_size; (void)ws_size;
  const int*   ids       = (const int*)d_in[0];
  const int*   dec_inp   = (const int*)d_in[1];
  const float* E         = (const float*)d_in[2];
  const float* e_law     = (const float*)d_in[3];
  const float* e_lab     = (const float*)d_in[4];
  const float* e_xw      = (const float*)d_in[5];
  const float* e_xb      = (const float*)d_in[6];
  const float* e_cw      = (const float*)d_in[7];
  const float* e_cb      = (const float*)d_in[8];
  const float* e_gw      = (const float*)d_in[9];
  const float* e_gb      = (const float*)d_in[10];
  const float* e_ng      = (const float*)d_in[11];
  const float* e_nb      = (const float*)d_in[12];
  const float* e_ow      = (const float*)d_in[13];
  const float* e_ob      = (const float*)d_in[14];
  const float* el_law    = (const float*)d_in[15];
  const float* el_lab    = (const float*)d_in[16];
  const float* el_xw     = (const float*)d_in[17];
  const float* el_xb     = (const float*)d_in[18];
  const float* el_cw     = (const float*)d_in[19];
  const float* el_cb     = (const float*)d_in[20];
  const float* ed_w      = (const float*)d_in[21];
  const float* ed_b      = (const float*)d_in[22];
  const float* d_law     = (const float*)d_in[23];
  const float* d_lab     = (const float*)d_in[24];
  const float* d_xw      = (const float*)d_in[25];
  const float* d_xb      = (const float*)d_in[26];
  const float* d_cw      = (const float*)d_in[27];
  const float* d_cb      = (const float*)d_in[28];
  const float* d_gw      = (const float*)d_in[29];
  const float* d_gb      = (const float*)d_in[30];
  const float* d_ng      = (const float*)d_in[31];
  const float* d_nb      = (const float*)d_in[32];
  const float* d_ow      = (const float*)d_in[33];
  const float* d_ob      = (const float*)d_in[34];
  const float* D_w       = (const float*)d_in[35];
  const float* D_b       = (const float*)d_in[36];
  float* out = (float*)d_out;

  char* w = (char*)d_ws;
  size_t off = 0;
  auto alloc = [&](size_t bytes) -> void* {
    void* p = w + off;
    off = (off + bytes + 255) & ~size_t(255);
    return p;
  };
  float*  y       = (float*)alloc(sizeof(float) * BSZ * TENC * DMODEL);
  __bf16* ybf     = (__bf16*)alloc(2ull * BSZ * TENC * DMODEL);
  float*  xbc     = (float*)alloc(sizeof(float) * BSZ * TENC * NH * CH_ENC);
  float*  XBC     = (float*)alloc(sizeof(float) * BSZ * TENC * NH * CH_ENC);
  float*  logA    = (float*)alloc(sizeof(float) * BSZ * TENC * NH);
  float*  sbuf    = (float*)alloc(sizeof(float) * BSZ * NH * TENC);
  __bf16* XbfT    = (__bf16*)alloc(2ull * BSZ * NH * CHD * TENC);
  __bf16* Bmbf    = (__bf16*)alloc(2ull * BSZ * NH * TENC * CHD);
  __bf16* Cbf     = (__bf16*)alloc(2ull * BSZ * NH * TENC * CHD);
  float*  Yb      = (float*)alloc(sizeof(float) * BSZ * TENC * DMODEL);
  float*  gbuf    = (float*)alloc(sizeof(float) * BSZ * TENC * DMODEL);
  __bf16* Ynbf    = (__bf16*)alloc(2ull * BSZ * TENC * DMODEL);
  float*  hsbuf   = (float*)alloc(sizeof(float) * LL * 32 * CHD * NENC);
  float*  Hsbuf   = (float*)alloc(sizeof(float) * LL * 32 * CHD * NDEC);
  float*  cache   = (float*)alloc(sizeof(float) * LL * BSZ * NH * CH_DEC * 3);
  float*  emb     = (float*)alloc(sizeof(float) * BSZ * TDEC * DMODEL);
  float*  yt      = (float*)alloc(sizeof(float) * BSZ * DMODEL);
  float*  xbc_dec = (float*)alloc(sizeof(float) * 32 * CH_DEC);
  float*  gdec    = (float*)alloc(sizeof(float) * 32 * CHD);
  float*  adec    = (float*)alloc(sizeof(float) * 32);
  float*  gY      = (float*)alloc(sizeof(float) * BSZ * DMODEL);
  float*  Yn_dec  = (float*)alloc(sizeof(float) * BSZ * DMODEL);
  __bf16* ytall   = (__bf16*)alloc(2ull * BSZ * TDEC * DMODEL);
  __bf16* w_xbc   = (__bf16*)alloc(2ull * (LL - 1) * NH * CH_ENC * CHD);
  __bf16* w_xbcl  = (__bf16*)alloc(2ull * NH * CH_ENCL * CHD);
  __bf16* w_gate  = (__bf16*)alloc(2ull * (LL - 1) * NH * CHD * CHD);
  __bf16* w_out   = (__bf16*)alloc(2ull * (LL - 1) * DMODEL * DMODEL);
  __bf16* w_D     = (__bf16*)alloc(2ull * VOCABN * DMODEL);

  // ---- prep: weight f32->bf16 (native (N,K) layout == WMMA B operand) ----
  {
    long long t1 = (long long)(LL - 1) * NH * CH_ENC * CHD;
    k_cvt<<<(int)((t1 + 255) / 256), 256, 0, stream>>>(e_xw, w_xbc, t1);
    long long t2 = (long long)NH * CH_ENCL * CHD;
    k_cvt<<<(int)((t2 + 255) / 256), 256, 0, stream>>>(el_xw, w_xbcl, t2);
    long long t3 = (long long)(LL - 1) * NH * CHD * CHD;
    k_cvt<<<(int)((t3 + 255) / 256), 256, 0, stream>>>(e_gw, w_gate, t3);
    long long t4 = (long long)(LL - 1) * DMODEL * DMODEL;
    k_cvt<<<(int)((t4 + 255) / 256), 256, 0, stream>>>(e_ow, w_out, t4);
    long long t5 = (long long)VOCABN * DMODEL;
    k_cvt<<<(int)((t5 + 255) / 256), 256, 0, stream>>>(D_w, w_D, t5);
  }
  k_zero<<<(LL * BSZ * NH * CH_DEC * 3 + 255) / 256, 256, 0, stream>>>(
      cache, LL * BSZ * NH * CH_DEC * 3);
  k_embed<<<BSZ * TENC, 256, 0, stream>>>(ids, E, y, ybf);
  k_embed<<<BSZ * TDEC, 256, 0, stream>>>(dec_inp, E, emb, (__bf16*)nullptr);

  // ---- encoder layers 0..6 ----
  for (int i = 0; i < LL - 1; ++i) {
    k_logA<<<BSZ * TENC * NH / 256, 256, 0, stream>>>(
        y, e_law + (long long)i * NH * CHD, e_lab + i * NH, logA, BSZ * TENC * NH);
    k_cumsum<<<1, 32, 0, stream>>>(logA, sbuf);
    // xbc projection: per (b,h)  M=256, N=192, K=64
    k_gemm<<<dim3(4 * 2, BSZ * NH), 256, 0, stream>>>(
        ybf, DMODEL, (long long)TENC * DMODEL, CHD,
        w_xbc + (long long)i * NH * CH_ENC * CHD, CHD, 0, (long long)CH_ENC * CHD,
        xbc, nullptr, NH * CH_ENC, (long long)TENC * NH * CH_ENC, CH_ENC,
        e_xb + (long long)i * NH * CH_ENC, CH_ENC, nullptr,
        TENC, CH_ENC, CHD, 2, NH);
    k_conv_silu<<<(BSZ * TENC * NH * CH_ENC + 255) / 256, 256, 0, stream>>>(
        xbc, e_cw + (long long)i * NH * CH_ENC * KK, e_cb + (long long)i * NH * CH_ENC,
        XBC, XbfT, Bmbf, Cbf, CH_ENC, 1, (long long)BSZ * TENC * NH * CH_ENC);
    k_attn<<<dim3(TENC / 64, BSZ * NH), 256, 0, stream>>>(Cbf, Bmbf, XbfT, sbuf, Yb);
    k_hT<<<dim3(16, BSZ * NH), 256, 0, stream>>>(
        XBC, sbuf, hsbuf + (long long)i * 32 * CHD * NENC, CH_ENC);
    // gate projection: per (b,h) M=256, N=64, K=64
    k_gemm<<<dim3(4 * 1, BSZ * NH), 256, 0, stream>>>(
        ybf, DMODEL, (long long)TENC * DMODEL, CHD,
        w_gate + (long long)i * NH * CHD * CHD, CHD, 0, (long long)CHD * CHD,
        gbuf, nullptr, DMODEL, (long long)TENC * DMODEL, CHD,
        e_gb + (long long)i * NH * CHD, CHD, nullptr,
        TENC, CHD, CHD, 1, NH);
    k_gate_ln<<<BSZ * TENC, 256, 0, stream>>>(gbuf, Yb, e_ng + i * DMODEL,
                                              e_nb + i * DMODEL, Ynbf);
    // out projection + residual M=512, N=1024, K=1024 (in-place on y / ybf)
    k_gemm<<<dim3(8 * 8, 1), 256, 0, stream>>>(
        Ynbf, DMODEL, 0, 0,
        w_out + (long long)i * DMODEL * DMODEL, DMODEL, 0, 0,
        y, ybf, DMODEL, 0, 0,
        e_ob + (long long)i * DMODEL, 0, y,
        BSZ * TENC, DMODEL, DMODEL, 8, 1);
  }
  // ---- last encoder layer (state only) ----
  k_logA<<<BSZ * TENC * NH / 256, 256, 0, stream>>>(y, el_law, el_lab, logA,
                                                    BSZ * TENC * NH);
  k_cumsum<<<1, 32, 0, stream>>>(logA, sbuf);
  k_gemm<<<dim3(4 * 1, BSZ * NH), 256, 0, stream>>>(
      ybf, DMODEL, (long long)TENC * DMODEL, CHD,
      w_xbcl, CHD, 0, (long long)CH_ENCL * CHD,
      xbc, nullptr, NH * CH_ENCL, (long long)TENC * NH * CH_ENCL, CH_ENCL,
      el_xb, CH_ENCL, nullptr, TENC, CH_ENCL, CHD, 1, NH);
  k_conv_silu<<<(BSZ * TENC * NH * CH_ENCL + 255) / 256, 256, 0, stream>>>(
      xbc, el_cw, el_cb, XBC, XbfT, Bmbf, Cbf, CH_ENCL, 0,
      (long long)BSZ * TENC * NH * CH_ENCL);
  k_hT<<<dim3(16, BSZ * NH), 256, 0, stream>>>(
      XBC, sbuf, hsbuf + (long long)(LL - 1) * 32 * CHD * NENC, CH_ENCL);

  // ---- encoder->decoder state projection ----
  k_ed<<<LL * 32 * CHD * NDEC / 256, 256, 0, stream>>>(hsbuf, ed_w, ed_b, Hsbuf);

  // ---- decoder recurrence (sequential; tiny M -> VALU matvecs) ----
  for (int t = 0; t < TDEC; ++t) {
    k_copy_yt<<<BSZ * DMODEL / 256, 256, 0, stream>>>(emb, yt, t);
    for (int l = 0; l < LL; ++l) {
      k_dec_proj<<<BSZ * NH, 256, 0, stream>>>(yt, d_xw, d_xb, d_gw, d_gb, d_law,
                                               d_lab, xbc_dec, gdec, adec, l);
      k_dec_update<<<BSZ * NH, 128, 0, stream>>>(xbc_dec, d_cw, d_cb, cache, Hsbuf,
                                                 gdec, adec, gY, l);
      k_dec_ln<<<BSZ, 256, 0, stream>>>(gY, d_ng + l * DMODEL, d_nb + l * DMODEL,
                                        Yn_dec);
      k_dec_out<<<BSZ * DMODEL / 256, 256, 0, stream>>>(
          Yn_dec, d_ow, d_ob, yt, ytall, l, t, (l == LL - 1) ? 1 : 0);
    }
  }

  // ---- vocab logits GEMM: M=256, N=20000, K=1024 (B = D_w native (N,K)) ----
  {
    int ntN = (VOCABN + 127) / 128;  // 157
    k_gemm<<<dim3(4 * ntN, 1), 256, 0, stream>>>(
        ytall, DMODEL, 0, 0, w_D, DMODEL, 0, 0,
        out, nullptr, VOCABN, 0, 0, D_b, 0, nullptr,
        BSZ * TDEC, VOCABN, DMODEL, ntN, 1);
  }
}
